// MHA_41987600285819
// MI455X (gfx1250) — matrix operand order repaired and checked by reading, not compiled, not run on hardware
//
#include <hip/hip_runtime.h>
#include <hip/hip_bf16.h>

// ---------------------------------------------------------------------------
// MHA forward for MI455X (gfx1250), bf16 WMMA pipeline, fp32 accumulate.
//   B=2, S=2048, D=2048, H=16, HD=128
// ---------------------------------------------------------------------------

#define BB 2
#define SS 2048
#define DD 2048
#define HH 16
#define HD 128
#define MM (BB * SS)   // 4096

typedef __attribute__((ext_vector_type(16))) __bf16   v16bf;
typedef __attribute__((ext_vector_type(8)))  float    v8f;
typedef __attribute__((ext_vector_type(4)))  unsigned uintx4;

// ---- bf16 helpers (round-to-nearest-even) ---------------------------------
__device__ __forceinline__ __bf16 f2bf(float x) {
    union { float f; unsigned u; } a; a.f = x;
    unsigned u = a.u + 0x7FFFu + ((a.u >> 16) & 1u);
    union { unsigned short s; __bf16 b; } r; r.s = (unsigned short)(u >> 16);
    return r.b;
}
__device__ __forceinline__ float bf2f(__bf16 x) {
    union { unsigned short s; __bf16 b; } a; a.b = x;
    union { float f; unsigned u; } r; r.u = ((unsigned)a.s) << 16;
    return r.f;
}
__device__ __forceinline__ unsigned pack2f(float a, float b) {
    union { float f; unsigned u; } x, y; x.f = a; y.f = b;
    unsigned ua = x.u + 0x7FFFu + ((x.u >> 16) & 1u);
    unsigned ub = y.u + 0x7FFFu + ((y.u >> 16) & 1u);
    return (ua >> 16) | (ub & 0xFFFF0000u);
}

// ---- fragment loaders (two / one contiguous 16B chunks) -------------------
__device__ __forceinline__ v16bf frag2(const __bf16* p0, const __bf16* p1) {
    union { v16bf v; uintx4 q[2]; } f;
    f.q[0] = *(const uintx4*)p0;
    f.q[1] = *(const uintx4*)p1;
    return f.v;
}
__device__ __forceinline__ v16bf frag1(const __bf16* p) {
    union { v16bf v; uintx4 q[2]; } f;
    f.q[0] = *(const uintx4*)p;
    f.q[1] = *(const uintx4*)(p + 8);
    return f.v;
}
__device__ __forceinline__ v8f wmma_bf16(v16bf a, v16bf b, v8f c) {
    return __builtin_amdgcn_wmma_f32_16x16x32_bf16(
        false, a, false, b, (short)0, c, false, false);
}

// ---------------------------------------------------------------------------
// Generic GEMM: C(M x 2048) = A(M x 2048, f32) * W(2048 x 2048, f32)
// MODE 0: write f32 row-major to out.
// MODE 1: write bf16 into (B,H,S,HD) layout (row = b*S+s, col = h*HD+d).
// Workgroup tile 128x128, K-step 64, 8 waves of 64x32 each
// -> 16 WMMAs per barrier pair.
// ---------------------------------------------------------------------------
#define AST 72   // padded LDS stride (halves) for 64-wide K tiles (144B, 16B-aligned)
template <int MODE>
__global__ __launch_bounds__(256) void gemm_bf16_kernel(
        const float* __restrict__ A, const float* __restrict__ W,
        void* __restrict__ outp)
{
    __shared__ __bf16 As[128 * AST];
    __shared__ __bf16 Bt[128 * AST];   // transposed: [n][k]

    const int tid  = threadIdx.x;
    const int lane = tid & 31;
    const int wave = tid >> 5;
    const int g    = lane >> 4;
    const int lm   = lane & 15;
    const int mBase = blockIdx.x * 128;
    const int nBase = blockIdx.y * 128;
    const int mOff  = (wave >> 2) * 64;   // 0 or 64
    const int nOff  = (wave & 3) * 32;    // 0..96

    const v8f zero = {0.f,0.f,0.f,0.f,0.f,0.f,0.f,0.f};
    v8f acc[4][2];
#pragma unroll
    for (int mt = 0; mt < 4; ++mt)
#pragma unroll
        for (int nt = 0; nt < 2; ++nt) acc[mt][nt] = zero;

    for (int kb = 0; kb < DD; kb += 64) {
        // ---- stage A tile 128x64 (f32 -> bf16), row-major padded ----------
#pragma unroll
        for (int it = 0; it < 8; ++it) {
            int idx = tid + it * 256;                  // 2048 float4 chunks
            int row = idx >> 4;                        // 16 float4 per row
            int c4  = (idx & 15) << 2;
            const float* src = A + (size_t)(mBase + row) * DD + kb + c4;
            float x0 = src[0], x1 = src[1], x2 = src[2], x3 = src[3];
            unsigned* dst = (unsigned*)&As[row * AST + c4];
            dst[0] = pack2f(x0, x1);
            dst[1] = pack2f(x2, x3);
        }
        // ---- stage W tile 64x128 transposed -> Bt[n][k] -------------------
#pragma unroll
        for (int it = 0; it < 8; ++it) {
            int idx = tid + it * 256;                  // 2048 float4 chunks
            int row = idx >> 5;                        // k within tile (0..63)
            int c4  = (idx & 31) << 2;                 // n within tile
            const float* src = W + (size_t)(kb + row) * DD + nBase + c4;
#pragma unroll
            for (int j = 0; j < 4; ++j)
                Bt[(c4 + j) * AST + row] = f2bf(src[j]);
        }
        // speculative prefetch of the next K-tile (global_prefetch_b8)
        if (kb + 64 < DD) {
            __builtin_prefetch(A + (size_t)(mBase + (tid & 127)) * DD + kb + 64, 0, 1);
            __builtin_prefetch(W + (size_t)(kb + 64 + (tid & 63)) * DD + nBase, 0, 1);
        }
        __syncthreads();

#pragma unroll
        for (int kk = 0; kk < 64; kk += 32) {
            v16bf aF[4], bF[2];
#pragma unroll
            for (int mt = 0; mt < 4; ++mt) {
                int m = mOff + mt * 16 + lm;
                aF[mt] = frag2(&As[m * AST + kk + 8 * g],
                               &As[m * AST + kk + 16 + 8 * g]);
            }
#pragma unroll
            for (int nt = 0; nt < 2; ++nt) {
                int n = nOff + nt * 16 + lm;
                bF[nt] = frag2(&Bt[n * AST + kk + 8 * g],
                               &Bt[n * AST + kk + 16 + 8 * g]);
            }
#pragma unroll
            for (int mt = 0; mt < 4; ++mt)
#pragma unroll
                for (int nt = 0; nt < 2; ++nt)
                    acc[mt][nt] = wmma_bf16(aF[mt], bF[nt], acc[mt][nt]);
        }
        __syncthreads();
    }

    // ---- epilogue (compile-time mode, no branches) ------------------------
#pragma unroll
    for (int mt = 0; mt < 4; ++mt) {
#pragma unroll
        for (int nt = 0; nt < 2; ++nt) {
            int col = nBase + nOff + nt * 16 + lm;
#pragma unroll
            for (int e = 0; e < 8; ++e) {
                int row = mBase + mOff + mt * 16 + e + 8 * g;
                float v = acc[mt][nt][e];
                if (MODE == 0) {
                    ((float*)outp)[(size_t)row * DD + col] = v;
                } else {
                    int b = row >> 11, s = row & (SS - 1);
                    int h = col >> 7,  d = col & (HD - 1);
                    ((__bf16*)outp)[(((size_t)(b * HH + h)) * SS + s) * HD + d]
                        = f2bf(v);
                }
            }
        }
    }
}

// ---------------------------------------------------------------------------
// RoPE applied in-place to q and v (the reference rotates q and v).
// One thread per (b,h,s,pair). Tensors are bf16 (B,H,S,HD).
// ---------------------------------------------------------------------------
__global__ __launch_bounds__(256) void rope_kernel(
        __bf16* __restrict__ q, __bf16* __restrict__ v,
        const float* __restrict__ cosT, const float* __restrict__ sinT)
{
    size_t p = (size_t)blockIdx.x * 256 + threadIdx.x;  // BB*HH*SS*64 total
    int i  = (int)(p & 63);
    int s  = (int)((p >> 6) & (SS - 1));
    int bh = (int)(p >> 17);
    size_t base = ((size_t)bh * SS + s) * HD + 2 * i;
    float c  = cosT[s * 64 + i];
    float sn = sinT[s * 64 + i];
    {
        float te = bf2f(q[base]), to = bf2f(q[base + 1]);
        q[base]     = f2bf(te * c - to * sn);
        q[base + 1] = f2bf(te * sn + to * c);
    }
    {
        float te = bf2f(v[base]), to = bf2f(v[base + 1]);
        v[base]     = f2bf(te * c - to * sn);
        v[base + 1] = f2bf(te * sn + to * c);
    }
}

// ---------------------------------------------------------------------------
// Flash attention, causal. One workgroup per (128 q rows, b*h).
// Each wave owns 16 q rows, iterates K in blocks of 32.
// Q,K,V bf16 in (B,H,S,HD); output f32 (B,S,D).
// ---------------------------------------------------------------------------
#define KST 136   // padded LDS stride (halves) for 128-wide rows
#define VST 40    // padded LDS stride for Vt / Ps rows (32-wide K)
__global__ __launch_bounds__(256) void fa_kernel(
        const __bf16* __restrict__ Q, const __bf16* __restrict__ Kg,
        const __bf16* __restrict__ V, float* __restrict__ Out)
{
    __shared__ __bf16 Ks[32 * KST];       // [k][d] row-major, padded
    __shared__ __bf16 Vt[128 * VST];      // [d][k] transposed, padded
    __shared__ __bf16 Ps[8][16 * VST];    // per-wave P tile [q][k]

    const int tid  = threadIdx.x;
    const int lane = tid & 31;
    const int wave = tid >> 5;
    const int g    = lane >> 4;
    const int lm   = lane & 15;
    const int qc   = blockIdx.x;          // 0..15
    const int bh   = blockIdx.y;          // 0..31
    const int qBase0 = qc * 128;
    const int qBase  = qBase0 + wave * 16;

    const __bf16* Qb = Q  + (size_t)bh * SS * HD;
    const __bf16* Kb = Kg + (size_t)bh * SS * HD;
    const __bf16* Vb = V  + (size_t)bh * SS * HD;

    // Q fragments for this wave's 16 rows (loaded once from global).
    v16bf qF[4];
#pragma unroll
    for (int d4 = 0; d4 < 4; ++d4) {
        const __bf16* r = Qb + (size_t)(qBase + lm) * HD + d4 * 32;
        qF[d4] = frag2(r + 8 * g, r + 16 + 8 * g);
    }

    const v8f zero = {0.f,0.f,0.f,0.f,0.f,0.f,0.f,0.f};
    v8f Oacc[8];
    float mrow[8], lrow[8];
#pragma unroll
    for (int dt = 0; dt < 8; ++dt) Oacc[dt] = zero;
#pragma unroll
    for (int e = 0; e < 8; ++e) { mrow[e] = -3.0e38f; lrow[e] = 0.f; }

    // scale = log2(e) / sqrt(HD); softmax done in base-2 space.
    const float kscale = 1.4426950408889634f * 0.08838834764831845f;
    const int kEnd = qBase0 + 128;

    for (int kb = 0; kb < kEnd; kb += 32) {
        __syncthreads();
        // ---- stage K block 32x128 row-major (padded) ----------------------
        {
            int row = tid >> 3;            // 0..31
            int seg = (tid & 7) * 16;      // 0..112
            const uintx4* src = (const uintx4*)(Kb + (size_t)(kb + row) * HD + seg);
            uintx4* dst = (uintx4*)&Ks[row * KST + seg];
            dst[0] = src[0];
            dst[1] = src[1];
        }
        // ---- stage V block transposed Vt[d][k] ----------------------------
        {
            int k  = tid & 31;
            int dB = (tid >> 5) * 16;
            const __bf16* src = Vb + (size_t)(kb + k) * HD + dB;
#pragma unroll
            for (int j = 0; j < 16; ++j)
                Vt[(dB + j) * VST + k] = src[j];
        }
        // prefetch next K/V block while this one is consumed
        if (kb + 32 < kEnd) {
            __builtin_prefetch(Kb + (size_t)(kb + 32 + (tid & 31)) * HD, 0, 1);
            __builtin_prefetch(Vb + (size_t)(kb + 32 + (tid & 31)) * HD, 0, 1);
        }
        __syncthreads();

        if (kb <= qBase + 15) {
            // ---- scores: two 16x16 tiles over 32 k-columns ----------------
            v8f s0 = zero, s1 = zero;
#pragma unroll
            for (int d4 = 0; d4 < 4; ++d4) {
                v16bf b0 = frag2(&Ks[(lm)      * KST + d4 * 32 + 8 * g],
                                 &Ks[(lm)      * KST + d4 * 32 + 16 + 8 * g]);
                v16bf b1 = frag2(&Ks[(16 + lm) * KST + d4 * 32 + 8 * g],
                                 &Ks[(16 + lm) * KST + d4 * 32 + 16 + 8 * g]);
                s0 = wmma_bf16(qF[d4], b0, s0);
                s1 = wmma_bf16(qF[d4], b1, s1);
            }
            const int kc0 = kb + lm;
            const int kc1 = kb + 16 + lm;
            // ---- online softmax per row (rows live per 16-lane half) ------
#pragma unroll
            for (int e = 0; e < 8; ++e) {
                int qi = qBase + e + 8 * g;
                float y0 = (kc0 <= qi) ? s0[e] * kscale : -3.0e38f;
                float y1 = (kc1 <= qi) ? s1[e] * kscale : -3.0e38f;
                float mx = fmaxf(y0, y1);
#pragma unroll
                for (int off = 1; off < 16; off <<= 1)
                    mx = fmaxf(mx, __shfl_xor(mx, off, 32));
                float mnew  = fmaxf(mrow[e], mx);
                float alpha = exp2f(mrow[e] - mnew);
                float p0 = exp2f(y0 - mnew);
                float p1 = exp2f(y1 - mnew);
                float ps = p0 + p1;
#pragma unroll
                for (int off = 1; off < 16; off <<= 1)
                    ps += __shfl_xor(ps, off, 32);
                lrow[e] = lrow[e] * alpha + ps;
                mrow[e] = mnew;
#pragma unroll
                for (int dt = 0; dt < 8; ++dt) Oacc[dt][e] *= alpha;
                // park P in wave-private LDS in row-major [q][k]
                Ps[wave][(e + 8 * g) * VST + lm]      = f2bf(p0);
                Ps[wave][(e + 8 * g) * VST + 16 + lm] = f2bf(p1);
            }
            // wave-private LDS RAW: wait DS ops, no workgroup barrier needed
            asm volatile("s_wait_dscnt 0x0" ::: "memory");
            v16bf pF = frag2(&Ps[wave][lm * VST + 8 * g],
                             &Ps[wave][lm * VST + 16 + 8 * g]);
#pragma unroll
            for (int dt = 0; dt < 8; ++dt) {
                v16bf bV = frag2(&Vt[(dt * 16 + lm) * VST + 8 * g],
                                 &Vt[(dt * 16 + lm) * VST + 16 + 8 * g]);
                Oacc[dt] = wmma_bf16(pF, bV, Oacc[dt]);
            }
        }
    }

    // ---- finalize: divide by l, write f32 (B,S,D) -------------------------
    const int b = bh >> 4, h = bh & (HH - 1);
#pragma unroll
    for (int e = 0; e < 8; ++e) {
        float rl = 1.0f / lrow[e];
        int qi = qBase + e + 8 * g;
        float* orow = Out + ((size_t)(b * SS + qi)) * DD + h * HD;
#pragma unroll
        for (int dt = 0; dt < 8; ++dt)
            orow[dt * 16 + lm] = Oacc[dt][e] * rl;
    }
}

// ---------------------------------------------------------------------------
// Launch: qkv GEMMs -> RoPE(q,v) -> flash attention -> output GEMM.
// ---------------------------------------------------------------------------
extern "C" void kernel_launch(void* const* d_in, const int* in_sizes, int n_in,
                              void* d_out, int out_size, void* d_ws, size_t ws_size,
                              hipStream_t stream) {
    const float* x    = (const float*)d_in[0];
    const float* cosT = (const float*)d_in[1];
    const float* sinT = (const float*)d_in[2];
    const float* wq   = (const float*)d_in[3];
    const float* wk   = (const float*)d_in[4];
    const float* wv   = (const float*)d_in[5];
    const float* wo   = (const float*)d_in[6];

    const size_t elems = (size_t)BB * SS * DD;   // 8,388,608
    __bf16* qb = (__bf16*)d_ws;
    __bf16* kb = qb + elems;
    __bf16* vb = kb + elems;
    float*  attn = (float*)(vb + elems);

    dim3 blk(256);
    dim3 gGrid(MM / 128, DD / 128);              // 32 x 16

    gemm_bf16_kernel<1><<<gGrid, blk, 0, stream>>>(x, wq, (void*)qb);
    gemm_bf16_kernel<1><<<gGrid, blk, 0, stream>>>(x, wk, (void*)kb);
    gemm_bf16_kernel<1><<<gGrid, blk, 0, stream>>>(x, wv, (void*)vb);

    rope_kernel<<<(BB * HH * SS * 64) / 256, blk, 0, stream>>>(qb, vb, cosT, sinT);

    fa_kernel<<<dim3(SS / 128, BB * HH), blk, 0, stream>>>(qb, kb, vb, attn);

    gemm_bf16_kernel<0><<<gGrid, blk, 0, stream>>>(attn, wo, d_out);
}
